// SimpleRNN_23828478558367
// MI455X (gfx1250) — matrix operand order repaired and checked by reading, not compiled, not run on hardware
//
#include <hip/hip_runtime.h>
#include <hip/hip_bf16.h>

// ---------------------------------------------------------------------------
// SimpleRNN forward on gfx1250 (MI455X), wave32 + WMMA bf16.
//   B=64, T=512, EMB=256, HID=512, OUT=3, VOCAB=50000
// Pipeline:
//   0) convert weights to bf16 (transposed for B-fragment loads), init h0/cnt
//   1) gather embedding rows -> bf16 x buffer (t-major rows)
//   2) input projection GEMM (32768x256)x(256x512) via WMMA, pure loads+wmma
//   3) persistent recurrent scan, 16 WGs, W_hh^T slice in LDS via async copy
//   4) tiny final dense
// ---------------------------------------------------------------------------

typedef __attribute__((ext_vector_type(16))) __bf16 v16bf;
typedef __attribute__((ext_vector_type(8)))  float  v8f;

union VBF {
    v16bf v;
    unsigned short u[16];
    uint4 q[2];
};

__device__ __forceinline__ unsigned short f2bf(float f) {
    __bf16 h = (__bf16)f;                       // hardware RTNE convert
    return __builtin_bit_cast(unsigned short, h);
}

__device__ __forceinline__ float bf2f(unsigned short b) {
    return __uint_as_float(((unsigned)b) << 16);
}

// tanh via HW transcendentals: 1 - 2/(exp2(2*log2e*x)+1); saturates to +-1.
__device__ __forceinline__ float fast_tanh(float x) {
    float t = __builtin_amdgcn_exp2f(x * 2.885390081777927f); // 2*log2(e)
    return 1.0f - 2.0f * __builtin_amdgcn_rcpf(t + 1.0f);
}

// ---------------------------------------------------------------------------
// Kernel 0: one-time conversion/init.
// ---------------------------------------------------------------------------
__global__ __launch_bounds__(256) void rnn_convert(
    const float* __restrict__ W_xh, const float* __restrict__ W_hh,
    unsigned short* __restrict__ wxhT, unsigned short* __restrict__ whhT,
    unsigned short* __restrict__ h0, unsigned* __restrict__ cnt)
{
    int idx = blockIdx.x * 256 + threadIdx.x;   // grid covers 262144
    if (idx < 131072) {
        int n = idx >> 8, k = idx & 255;
        wxhT[idx] = f2bf(W_xh[k * 512 + n]);    // wxhT[n][k] = W_xh[k][n]
    }
    {
        int n = idx >> 9, k = idx & 511;
        whhT[idx] = f2bf(W_hh[k * 512 + n]);    // whhT[n][k] = W_hh[k][n]
    }
    if (idx < 32768) h0[idx] = 0;
    if (idx == 0) *cnt = 0u;
}

// ---------------------------------------------------------------------------
// Kernel 1: embedding gather -> bf16, rows in t-major order (r = t*64 + b).
// Each thread converts 4 contiguous elements (float4 in, 4x bf16 out).
// ---------------------------------------------------------------------------
__global__ __launch_bounds__(256) void rnn_gather(
    const int* __restrict__ tokens, const float* __restrict__ emb,
    unsigned short* __restrict__ xbf)
{
    int idx = blockIdx.x * 256 + threadIdx.x;   // 32768 rows * 64 chunks
    int r = idx >> 6;                           // row  (= t*64 + b)
    int c = idx & 63;                           // 4-elt chunk within EMB=256
    int b = r & 63;
    int t = r >> 6;
    int tok = tokens[b * 512 + t];              // tokens is [B,T] row-major
    float4 f = *(const float4*)(emb + (size_t)tok * 256 + c * 4);
    unsigned short o[4] = { f2bf(f.x), f2bf(f.y), f2bf(f.z), f2bf(f.w) };
    *(uint2*)(xbf + (size_t)r * 256 + c * 4) = *(const uint2*)o;
}

// ---------------------------------------------------------------------------
// Kernel 2: input projection.  xp[r][n] = xbf[r][:] . wxhT[n][:] + b_h[n]
// One 16x16 tile per wave; fully unrolled K-loop: 8 x wmma, pure b128 loads.
// ---------------------------------------------------------------------------
__global__ __launch_bounds__(256) void rnn_embed_proj(
    const unsigned short* __restrict__ xbf,
    const unsigned short* __restrict__ wxhT, const float* __restrict__ b_h,
    float* __restrict__ xp)
{
    int lane  = threadIdx.x & 31;
    int wave  = threadIdx.x >> 5;
    int tile  = blockIdx.x * 8 + wave;      // 65536 tiles = 2048 Mtiles x 32 Ntiles
    int mtile = tile >> 5;
    int ntile = tile & 31;
    int hsel  = (lane < 16) ? 0 : 1;        // which K-half this lane owns
    int l15   = lane & 15;

    const unsigned short* arow = xbf + (size_t)(mtile * 16 + l15) * 256;
    int n = ntile * 16 + l15;

    v8f acc = {};
    #pragma unroll
    for (int kk = 0; kk < 8; ++kk) {
        int k0 = kk * 32;
        VBF a, bb;
        // A fragment: 16 contiguous K starting at k0+hsel*16 (matches B)
        a.q[0] = *(const uint4*)(arow + k0 + hsel * 16);
        a.q[1] = *(const uint4*)(arow + k0 + hsel * 16 + 8);
        // B fragment: column n, 16 contiguous K (pre-transposed weights)
        const uint4* bp = (const uint4*)(wxhT + (size_t)n * 256 + k0 + hsel * 16);
        bb.q[0] = bp[0];
        bb.q[1] = bp[1];
        acc = __builtin_amdgcn_wmma_f32_16x16x32_bf16(
            false, a.v, false, bb.v, (short)0, acc, false, false);
    }
    float bias = b_h[n];
    #pragma unroll
    for (int i = 0; i < 8; ++i) {
        int rr = mtile * 16 + i + hsel * 8;          // C layout: M = i (+8 upper half)
        xp[(size_t)rr * 512 + n] = acc[i] + bias;
    }
}

// ---------------------------------------------------------------------------
// Kernel 3: persistent recurrent scan, 16 co-resident workgroups.
//   h_{t+1} = tanh(xp[t] + h_t @ W_hh)
// Each WG owns 32 HID columns; its W_hh^T slice (32KB) is async-copied into
// LDS once (global_load_async_to_lds_b128 + s_wait_asynccnt) and reused for
// all 512 steps. Grid barrier = monotonically increasing atomic counter.
// ---------------------------------------------------------------------------
#define NWG 16

__global__ __launch_bounds__(256) void rnn_scan(
    const unsigned short* __restrict__ whhT, const float* __restrict__ xp,
    unsigned short* __restrict__ h0, unsigned short* __restrict__ h1,
    unsigned* __restrict__ cnt)
{
    __shared__ unsigned short lws[32 * 512];        // this WG's W_hh^T slice, 32 KB

    int wg = blockIdx.x;
    int n0 = wg * 32;

    // One-time async copy: 2048 x 16B chunks, 8 per thread (ASYNCcnt path).
    {
        unsigned lds_base = (unsigned)(size_t)(&lws[0]);   // addr[31:0] = LDS offset
        unsigned long long gbase = (unsigned long long)(whhT + (size_t)n0 * 512);
        #pragma unroll
        for (int i = 0; i < 8; ++i) {
            unsigned chunk = threadIdx.x + i * 256;        // 0..2047
            unsigned ldsb = lds_base + chunk * 16;
            unsigned long long g = gbase + (unsigned long long)chunk * 16;
            asm volatile("global_load_async_to_lds_b128 %0, %1, off"
                         :: "v"(ldsb), "v"(g) : "memory");
        }
        asm volatile("s_wait_asynccnt 0x0" ::: "memory");
    }
    __syncthreads();

    int lane  = threadIdx.x & 31;
    int wave  = threadIdx.x >> 5;
    int mtile = wave & 3;                   // 4 M-tiles cover B=64
    int ntile = wave >> 2;                  // 2 N-tiles cover the 32-col slice
    int hsel  = (lane < 16) ? 0 : 1;
    int l15   = lane & 15;
    int nloc  = ntile * 16 + l15;           // 0..31 within slice
    int nglob = n0 + nloc;
    int rowA  = mtile * 16 + l15;           // hidden-state row this lane loads

    unsigned target = 0;

    for (int t = 0; t < 512; ++t) {
        const unsigned short* hc = (t & 1) ? h1 : h0;
        unsigned short*       hn = (t & 1) ? h0 : h1;

        v8f acc = {};
        #pragma unroll
        for (int kk = 0; kk < 16; ++kk) {
            int k0 = kk * 32;
            VBF a, bb;
            // A: h_t row, 16 contiguous K starting at k0+hsel*16 (matches B)
            a.q[0] = *(const uint4*)(hc + (size_t)rowA * 512 + k0 + hsel * 16);
            a.q[1] = *(const uint4*)(hc + (size_t)rowA * 512 + k0 + hsel * 16 + 8);
            // B: column nloc from LDS, 16 contiguous K (32B)
            const uint4* bp = (const uint4*)(lws + (size_t)nloc * 512 + k0 + hsel * 16);
            bb.q[0] = bp[0];
            bb.q[1] = bp[1];
            acc = __builtin_amdgcn_wmma_f32_16x16x32_bf16(
                false, a.v, false, bb.v, (short)0, acc, false, false);
        }

        // epilogue: add xp, tanh, store bf16 h_{t+1}
        const float* xpt = xp + (size_t)t * 64 * 512;
        #pragma unroll
        for (int i = 0; i < 8; ++i) {
            int brow = mtile * 16 + i + hsel * 8;
            float v = fast_tanh(acc[i] + xpt[(size_t)brow * 512 + nglob]);
            hn[(size_t)brow * 512 + nglob] = f2bf(v);
        }

        // prefetch next timestep's xp rows toward this WGP
        if (t + 1 < 512)
            __builtin_prefetch(xp + (size_t)(t + 1) * 64 * 512 + (size_t)rowA * 512, 0, 0);

        // ---- grid barrier (monotonic counter, no reset) ----
        target += NWG;
        __syncthreads();
        if (threadIdx.x == 0) {
            __threadfence();
            atomicAdd(cnt, 1u);
            while (atomicAdd(cnt, 0u) < target) {
                __builtin_amdgcn_s_sleep(2);
            }
        }
        __syncthreads();
        __threadfence();
    }
}

// ---------------------------------------------------------------------------
// Kernel 4: final dense  out[b][o] = h_last[b] . W_fc[:,o] + b_fc[o]
// T even -> final h lives in h0 buffer.
// ---------------------------------------------------------------------------
__global__ __launch_bounds__(256) void rnn_final(
    const unsigned short* __restrict__ hfin, const float* __restrict__ W_fc,
    const float* __restrict__ b_fc, float* __restrict__ out)
{
    int idx = threadIdx.x;
    if (idx >= 192) return;
    int b = idx / 3, o = idx % 3;
    float s = b_fc[o];
    for (int k = 0; k < 512; ++k)
        s += bf2f(hfin[b * 512 + k]) * W_fc[k * 3 + o];
    out[idx] = s;
}

// ---------------------------------------------------------------------------
// Host launcher
// ---------------------------------------------------------------------------
extern "C" void kernel_launch(void* const* d_in, const int* in_sizes, int n_in,
                              void* d_out, int out_size, void* d_ws, size_t ws_size,
                              hipStream_t stream) {
    const int*   tokens = (const int*)  d_in[0];   // [64,512] int32
    const float* emb    = (const float*)d_in[1];   // [50000,256]
    const float* W_xh   = (const float*)d_in[2];   // [256,512]
    const float* W_hh   = (const float*)d_in[3];   // [512,512]
    const float* b_h    = (const float*)d_in[4];   // [512]
    const float* W_fc   = (const float*)d_in[5];   // [512,3]
    const float* b_fc   = (const float*)d_in[6];   // [3]
    float*       out    = (float*)d_out;           // [64,3]

    char* ws = (char*)d_ws;
    size_t off = 0;
    float*          xp   = (float*)(ws + off);          off += (size_t)512 * 64 * 512 * 4; // 64 MB
    unsigned short* xbf  = (unsigned short*)(ws + off); off += (size_t)32768 * 256 * 2;    // 16 MB
    unsigned short* wxhT = (unsigned short*)(ws + off); off += (size_t)512 * 256 * 2;      // 256 KB
    unsigned short* whhT = (unsigned short*)(ws + off); off += (size_t)512 * 512 * 2;      // 512 KB
    unsigned short* h0   = (unsigned short*)(ws + off); off += (size_t)64 * 512 * 2;       // 64 KB
    unsigned short* h1   = (unsigned short*)(ws + off); off += (size_t)64 * 512 * 2;       // 64 KB
    unsigned*       cnt  = (unsigned*)(ws + off);

    rnn_convert<<<1024, 256, 0, stream>>>(W_xh, W_hh, wxhT, whhT, h0, cnt);
    rnn_gather<<<8192, 256, 0, stream>>>(tokens, emb, xbf);
    rnn_embed_proj<<<8192, 256, 0, stream>>>(xbf, wxhT, b_h, xp);
    rnn_scan<<<NWG, 256, 0, stream>>>(whhT, xp, h0, h1, cnt);
    rnn_final<<<1, 256, 0, stream>>>(h0, W_fc, b_fc, out);
}
